// CausalSelfAttention_73443940761853
// MI455X (gfx1250) — compile-verified, matrix operand order checked by reading, not run on hardware
//
#include <hip/hip_runtime.h>

// ---------------------------------------------------------------------------
// CDNA5 (gfx1250) causal self-attention:
//   k1: qkv = x @ w_attn   (f16 WMMA, fp32 acc) -> fp32 k/v to d_out,
//       f16 q (pre-scaled) / k / v^T to workspace
//   k2: flash attention over f16 q/k/vT, async-LDS double-buffered K/V
//   k3: y = y_att @ w_proj (f16 WMMA) -> fp32 y to d_out
// ---------------------------------------------------------------------------

typedef __attribute__((ext_vector_type(16))) _Float16 v16h;
typedef __attribute__((ext_vector_type(8)))  float    v8f;

#define B_  4
#define T_  2048
#define C_  1024
#define H_  16
#define D_  64

struct Frag16 {
    union { v16h v; unsigned u[8]; };
};

// Fragment gather for v_wmma_*_16x16x32_f16 operands (CDNA5 ISA 16-bit
// layout): lane half (lane>>4) adds K offset +8; dword p holds the K pair
// { (p&3)*2 + (p>>2)*16 (+8 for upper half-wave), +1 }. With a 16B-aligned
// row base this folds into two ds_load_b128 per fragment.
__device__ __forceinline__ v16h load_frag(const _Float16* rowBase, int half) {
    Frag16 f;
#pragma unroll
    for (int p = 0; p < 8; ++p) {
        int off = ((p & 3) << 1) + (half << 3) + ((p >> 2) << 4);
        f.u[p] = *reinterpret_cast<const unsigned*>(rowBase + off);
    }
    return f.v;
}

// --- CDNA5 async global->LDS copy (ASYNCcnt), 16B per lane -----------------
// GVS mode: 64-bit SGPR base + per-lane 32-bit byte offset; LDS dest address
// is the low 32 bits of the flat pointer (LDS aperture keeps offset there).
__device__ __forceinline__ void async_copy16(void* ldsDst, const void* gBase,
                                             unsigned gOffBytes) {
    unsigned ldsAddr = (unsigned)(unsigned long long)ldsDst;
    asm volatile("global_load_async_to_lds_b128 %0, %1, %2"
                 :
                 : "v"(ldsAddr), "v"(gOffBytes), "s"(gBase)
                 : "memory");
}
__device__ __forceinline__ void wait_async0() {
    asm volatile("s_wait_asynccnt 0x0" ::: "memory");
}

// ---------------------------------------------------------------------------
// GEMM out[M,N] = A[M,K] * Bw[K,N], f16 WMMA / fp32 acc.
// Block tile 128x128x32; 8 waves (4M x 2N); wave tile 32x64 = 8 accumulators.
// Register-prefetch pipeline + L2 prefetch 2 K-steps ahead.
// MODE 0: A f16 raw staging, plain fp32 store. MODE 1: A fp32 x, qkv split.
// ---------------------------------------------------------------------------
template <int MODE>
__global__ __launch_bounds__(256)
void wmma_gemm_kernel(const void* __restrict__ Aptr, const float* __restrict__ Bw,
                      float* __restrict__ outP,
                      _Float16* __restrict__ q16, float* __restrict__ outK,
                      _Float16* __restrict__ k16, float* __restrict__ outV,
                      _Float16* __restrict__ vt16,
                      int K, int lda, int ldb, int ldc) {
    __shared__ _Float16 As[128 * 32];   // [m][k]
    __shared__ _Float16 Bs[128 * 32];   // [n][k] transposed (pair-contiguous K)

    const _Float16* A16 = (const _Float16*)Aptr;
    const float*    Af  = (const float*)Aptr;

    const int tid  = threadIdx.x;
    const int lane = tid & 31;
    const int wid  = tid >> 5;
    const int wm   = wid & 3;
    const int wn   = wid >> 2;
    const int half = lane >> 4;
    const int lo   = lane & 15;
    const int m0   = blockIdx.y * 128;
    const int n0   = blockIdx.x * 128;

    uint4  aReg[2];     // MODE 0 raw f16 prefetch
    float4 aRegF[4];    // MODE 1 fp32 prefetch
    float4 bReg[4];

    auto fetchA = [&](int k0) {
        if constexpr (MODE == 0) {
#pragma unroll
            for (int i = 0; i < 2; ++i) {           // 128 rows x 4 16B chunks
                int c = tid + i * 256;
                int r = c >> 2, ch = c & 3;
                aReg[i] = *(const uint4*)(A16 + (size_t)(m0 + r) * lda + k0 + ch * 8);
            }
        } else {
#pragma unroll
            for (int i = 0; i < 4; ++i) {           // 128 rows x 8 float4 chunks
                int c = tid + i * 256;
                int r = c >> 3, ch = c & 7;
                aRegF[i] = *(const float4*)(Af + (size_t)(m0 + r) * lda + k0 + ch * 4);
            }
        }
    };
    auto fetchB = [&](int k0) {
#pragma unroll
        for (int i = 0; i < 4; ++i) {               // 32 k-rows x 32 float4 chunks
            int c = tid + i * 256;
            int kk = c >> 5, n4 = c & 31;
            bReg[i] = *(const float4*)(Bw + (size_t)(k0 + kk) * ldb + n0 + n4 * 4);
        }
    };
    auto prefetchL2 = [&](int k0) {                 // warm L2 two K-steps ahead
        if constexpr (MODE == 0)
            __builtin_prefetch(A16 + (size_t)(m0 + (tid >> 2)) * lda + k0, 0, 0);
        else
            __builtin_prefetch(Af + (size_t)(m0 + (tid >> 3)) * lda + k0, 0, 0);
        __builtin_prefetch(Bw + (size_t)(k0 + (tid >> 5)) * ldb + n0 + (tid & 31) * 4, 0, 0);
    };
    auto stage = [&]() {
        if constexpr (MODE == 0) {
#pragma unroll
            for (int i = 0; i < 2; ++i) {
                int c = tid + i * 256;
                int r = c >> 2, ch = c & 3;
                *(uint4*)(As + r * 32 + ch * 8) = aReg[i];
            }
        } else {
#pragma unroll
            for (int i = 0; i < 4; ++i) {
                int c = tid + i * 256;
                int r = c >> 3, ch = c & 7;
                uint2 pk;
                _Float16* hp = (_Float16*)&pk;
                hp[0] = (_Float16)aRegF[i].x; hp[1] = (_Float16)aRegF[i].y;
                hp[2] = (_Float16)aRegF[i].z; hp[3] = (_Float16)aRegF[i].w;
                *(uint2*)(As + r * 32 + ch * 4) = pk;
            }
        }
#pragma unroll
        for (int i = 0; i < 4; ++i) {
            int c = tid + i * 256;
            int kk = c >> 5, n4 = c & 31;
            Bs[(n4 * 4 + 0) * 32 + kk] = (_Float16)bReg[i].x;
            Bs[(n4 * 4 + 1) * 32 + kk] = (_Float16)bReg[i].y;
            Bs[(n4 * 4 + 2) * 32 + kk] = (_Float16)bReg[i].z;
            Bs[(n4 * 4 + 3) * 32 + kk] = (_Float16)bReg[i].w;
        }
    };

    v8f acc[2][4];
#pragma unroll
    for (int mt = 0; mt < 2; ++mt)
#pragma unroll
        for (int nt = 0; nt < 4; ++nt)
#pragma unroll
            for (int e = 0; e < 8; ++e) acc[mt][nt][e] = 0.0f;

    fetchA(0);
    fetchB(0);
    for (int k0 = 0; k0 < K; k0 += 32) {
        __syncthreads();          // all waves done reading previous tile
        stage();
        __syncthreads();
        if (k0 + 32 < K) {        // prefetch next tile; first use is next stage()
            fetchA(k0 + 32);
            fetchB(k0 + 32);
        }
        if (k0 + 64 < K) prefetchL2(k0 + 64);
        v16h a[2];
#pragma unroll
        for (int mt = 0; mt < 2; ++mt)
            a[mt] = load_frag(As + (wm * 32 + mt * 16 + lo) * 32, half);
#pragma unroll
        for (int nt = 0; nt < 4; ++nt) {
            v16h b = load_frag(Bs + (wn * 64 + nt * 16 + lo) * 32, half);
#pragma unroll
            for (int mt = 0; mt < 2; ++mt)
                acc[mt][nt] = __builtin_amdgcn_wmma_f32_16x16x32_f16(
                    false, a[mt], false, b, (short)0, acc[mt][nt], false, false);
        }
    }

    // Epilogue. C/D layout: lane holds col N=lo, rows M = r + 8*half.
#pragma unroll
    for (int mt = 0; mt < 2; ++mt) {
#pragma unroll
        for (int nt = 0; nt < 4; ++nt) {
#pragma unroll
            for (int r = 0; r < 8; ++r) {
                int gm = m0 + wm * 32 + mt * 16 + half * 8 + r;
                int gn = n0 + wn * 64 + nt * 16 + lo;
                float val = acc[mt][nt][r];
                if constexpr (MODE == 0) {
                    outP[(size_t)gm * ldc + gn] = val;
                } else {
                    int part = gn >> 10, cc = gn & 1023;
                    int h = cc >> 6, d = cc & 63;
                    int b = gm >> 11, t = gm & (T_ - 1);
                    size_t idx = ((size_t)(b * H_ + h) * T_ + t) * D_ + d;
                    if (part == 0) {
                        q16[idx] = (_Float16)(val * 0.125f);      // fold 1/sqrt(Dh)
                    } else if (part == 1) {
                        outK[idx] = val;
                        k16[idx]  = (_Float16)val;
                    } else {
                        outV[idx] = val;
                        vt16[((size_t)(b * H_ + h) * D_ + d) * T_ + t] = (_Float16)val;
                    }
                }
            }
        }
    }
}

// ---------------------------------------------------------------------------
// Flash attention (causal, online softmax), all-f16 inputs, fp32 softmax/acc.
// Block = (b, h, 128 q-rows); 8 waves x 16 q-rows; 64-key tiles.
// K/V staged by async global->LDS copies into ping-pong buffers: tile i+1
// streams in on the ASYNC engine while tile i runs on the WMMA pipe.
// One barrier per key tile.
// ---------------------------------------------------------------------------
__global__ __launch_bounds__(256)
void flash_attn_kernel(const _Float16* __restrict__ q16,
                       const _Float16* __restrict__ k16,
                       const _Float16* __restrict__ vt16,
                       _Float16* __restrict__ yatt16) {
    __shared__ _Float16 Qs[128 * 64];        // [qr][d]
    __shared__ _Float16 Ksb[2][64 * 64];     // [key][d]  (B-frag for Q*K^T)
    __shared__ _Float16 Vtb[2][64 * 64];     // [d][key]  (B-frag for P*V)
    __shared__ _Float16 Ps[8][16 * 64];      // per-wave P tile [row][key]

    const int tid  = threadIdx.x;
    const int lane = tid & 31;
    const int wid  = tid >> 5;
    const int half = lane >> 4;
    const int lo   = lane & 15;

    const int qb = blockIdx.x & (T_ / 128 - 1);
    const int bh = blockIdx.x / (T_ / 128);
    const int b  = bh >> 4;
    const int h  = bh & (H_ - 1);

    const _Float16* kBase = k16 + (size_t)bh * T_ * D_;
    const _Float16* vBase = vt16 + (size_t)bh * D_ * T_;
    const _Float16* qBase = q16 + (size_t)bh * T_ * D_;

    // Async-stage Q (f16, pre-scaled): 128 rows x 8 16B chunks
#pragma unroll
    for (int i = 0; i < 4; ++i) {
        int c = tid + i * 256;
        int r = c >> 3, ch = c & 7;
        async_copy16(Qs + r * 64 + ch * 8, qBase,
                     (unsigned)(((qb * 128 + r) * D_ + ch * 8) * sizeof(_Float16)));
    }

    auto stageKV = [&](int kt, int buf) {
#pragma unroll
        for (int i = 0; i < 2; ++i) {
            int c = tid + i * 256;             // 64 rows x 8 16B chunks
            int r = c >> 3, ch = c & 7;
            async_copy16(&Ksb[buf][r * 64 + ch * 8], kBase,
                         (unsigned)(((kt + r) * D_ + ch * 8) * sizeof(_Float16)));
            async_copy16(&Vtb[buf][r * 64 + ch * 8], vBase,
                         (unsigned)((r * T_ + kt + ch * 8) * sizeof(_Float16)));
        }
    };

    const int qRow0 = qb * 128 + wid * 16;
    float m_i[8], l_i[8];
    v8f o[4];
#pragma unroll
    for (int r = 0; r < 8; ++r) { m_i[r] = -1e30f; l_i[r] = 0.0f; }
#pragma unroll
    for (int dt = 0; dt < 4; ++dt)
#pragma unroll
        for (int e = 0; e < 8; ++e) o[dt][e] = 0.0f;

    const int nIter = (qb * 128 + 128) / 64;
    stageKV(0, 0);

    for (int it = 0; it < nIter; ++it) {
        const int kt  = it * 64;
        const int cur = it & 1;
        const _Float16* Ks = Ksb[cur];
        const _Float16* Vt = Vtb[cur];

        wait_async0();            // Q + tile `it` landed in LDS
        __syncthreads();          // ...and everyone's past reading buf 1-cur
        if (it + 1 < nIter) stageKV(kt + 64, 1 - cur);

        // S = Q * K^T : 4 key-subtiles x 2 d-slabs
        v16h aq[2];
#pragma unroll
        for (int dd = 0; dd < 2; ++dd)
            aq[dd] = load_frag(Qs + (wid * 16 + lo) * 64 + dd * 32, half);

        v8f sAcc[4];
#pragma unroll
        for (int kt2 = 0; kt2 < 4; ++kt2) {
#pragma unroll
            for (int e = 0; e < 8; ++e) sAcc[kt2][e] = 0.0f;
#pragma unroll
            for (int dd = 0; dd < 2; ++dd) {
                v16h bk = load_frag(Ks + (kt2 * 16 + lo) * 64 + dd * 32, half);
                sAcc[kt2] = __builtin_amdgcn_wmma_f32_16x16x32_f16(
                    false, aq[dd], false, bk, (short)0, sAcc[kt2], false, false);
            }
        }

        // Causal mask + online softmax (row lives in 16 lanes of a half-wave)
#pragma unroll
        for (int r = 0; r < 8; ++r) {
            const int gq = qRow0 + half * 8 + r;
            float s[4], p[4];
#pragma unroll
            for (int kt2 = 0; kt2 < 4; ++kt2) {
                int gk = kt + kt2 * 16 + lo;
                s[kt2] = (gk <= gq) ? sAcc[kt2][r] : -1e30f;
            }
            float mx = fmaxf(fmaxf(s[0], s[1]), fmaxf(s[2], s[3]));
#pragma unroll
            for (int d = 1; d < 16; d <<= 1)
                mx = fmaxf(mx, __shfl_xor(mx, d, 32));
            float mnew = fmaxf(m_i[r], mx);
            float rs = 0.0f;
#pragma unroll
            for (int kt2 = 0; kt2 < 4; ++kt2) {
                p[kt2] = __expf(s[kt2] - mnew);
                rs += p[kt2];
            }
#pragma unroll
            for (int d = 1; d < 16; d <<= 1)
                rs += __shfl_xor(rs, d, 32);
            float alpha = __expf(m_i[r] - mnew);
            l_i[r] = l_i[r] * alpha + rs;
            m_i[r] = mnew;
#pragma unroll
            for (int dt = 0; dt < 4; ++dt) o[dt][r] *= alpha;
#pragma unroll
            for (int kt2 = 0; kt2 < 4; ++kt2)
                Ps[wid][(half * 8 + r) * 64 + kt2 * 16 + lo] = (_Float16)p[kt2];
        }

        // O += P * V   (wave-private LDS round trip converts C->A layout)
        v16h ap[2];
#pragma unroll
        for (int kk = 0; kk < 2; ++kk)
            ap[kk] = load_frag(&Ps[wid][lo * 64 + kk * 32], half);
#pragma unroll
        for (int dt = 0; dt < 4; ++dt) {
#pragma unroll
            for (int kk = 0; kk < 2; ++kk) {
                v16h bv = load_frag(Vt + (dt * 16 + lo) * 64 + kk * 32, half);
                o[dt] = __builtin_amdgcn_wmma_f32_16x16x32_f16(
                    false, ap[kk], false, bv, (short)0, o[dt], false, false);
            }
        }
    }

    // Normalize; store f16 heads-merged activation for the proj GEMM
#pragma unroll
    for (int r = 0; r < 8; ++r) {
        float inv = 1.0f / l_i[r];
        int tg = qRow0 + half * 8 + r;
#pragma unroll
        for (int dt = 0; dt < 4; ++dt) {
            int c = h * D_ + dt * 16 + lo;
            yatt16[((size_t)b * T_ + tg) * C_ + c] = (_Float16)(o[dt][r] * inv);
        }
    }
}

// ---------------------------------------------------------------------------
// d_in:  [0]=x (B*T*C fp32), [1]=w_attn (C x 3C), [2]=w_proj (C x C)
// d_out: y | k | v  (fp32, k/v in [B,H,T,64])
// d_ws:  q16 | k16 | vT16 | yatt16  (f16, 16MB each = 64MB)
// ---------------------------------------------------------------------------
extern "C" void kernel_launch(void* const* d_in, const int* in_sizes, int n_in,
                              void* d_out, int out_size, void* d_ws, size_t ws_size,
                              hipStream_t stream) {
    (void)in_sizes; (void)n_in; (void)out_size; (void)ws_size;
    const float* x      = (const float*)d_in[0];
    const float* w_attn = (const float*)d_in[1];
    const float* w_proj = (const float*)d_in[2];

    const size_t NYel = (size_t)B_ * T_ * C_;     // 8388608
    float* y_out = (float*)d_out;
    float* k_out = y_out + NYel;
    float* v_out = y_out + 2 * NYel;

    _Float16* q16    = (_Float16*)d_ws;
    _Float16* k16    = q16 + NYel;
    _Float16* vt16   = k16 + NYel;
    _Float16* yatt16 = vt16 + NYel;

    // 1) qkv = x @ w_attn (split epilogue)
    {
        dim3 grid(3 * C_ / 128, (B_ * T_) / 128);
        wmma_gemm_kernel<1><<<grid, 256, 0, stream>>>(
            x, w_attn, nullptr, q16, k_out, k16, v_out, vt16,
            /*K=*/C_, /*lda=*/C_, /*ldb=*/3 * C_, /*ldc=*/0);
    }

    // 2) flash attention -> yatt16 [B,T,C] f16
    {
        dim3 grid(B_ * H_ * (T_ / 128));
        flash_attn_kernel<<<grid, 256, 0, stream>>>(q16, k16, vt16, yatt16);
    }

    // 3) y = yatt16 @ w_proj -> fp32 d_out
    {
        dim3 grid(C_ / 128, (B_ * T_) / 128);
        wmma_gemm_kernel<0><<<grid, 256, 0, stream>>>(
            yatt16, w_proj, y_out, nullptr, nullptr, nullptr, nullptr, nullptr,
            /*K=*/C_, /*lda=*/C_, /*ldb=*/C_, /*ldc=*/C_);
    }
}
